// Pointnet2SSG_20710332301422
// MI455X (gfx1250) — matrix operand order repaired and checked by reading, not compiled, not run on hardware
//
#include <hip/hip_runtime.h>
#include <hip/hip_bf16.h>

// ---------------------------------------------------------------------------
// PointNet++ SSG forward for gfx1250 (MI455X).
// GEMMs run through v_wmma_f32_16x16x32_f16: each 256-thread block owns a
// 128(M) x 64(N) macro-tile; the shared 64-col x 32-K W chunk is staged in
// LDS via a double-buffered Tensor Data Mover pipeline (cooperative loads as
// fallback); each wave holds 4 accumulators so one A fragment feeds 4
// back-to-back WMMAs whose B fragments come from LDS (ds_load_b128).
// ---------------------------------------------------------------------------

typedef _Float16 f16;
typedef __attribute__((ext_vector_type(16))) _Float16 v16h;
typedef __attribute__((ext_vector_type(8)))  _Float16 v8h;
typedef __attribute__((ext_vector_type(8)))  float    v8f;
typedef __attribute__((ext_vector_type(4)))  unsigned int u32x4;
typedef __attribute__((ext_vector_type(8)))  int i32x8;
typedef __attribute__((ext_vector_type(4)))  int i32x4;

#define BN_EPS 1e-5f

#if defined(__HIP_DEVICE_COMPILE__) && \
    __has_builtin(__builtin_amdgcn_tensor_load_to_lds) && \
    __has_builtin(__builtin_amdgcn_s_wait_tensorcnt)
#define HAVE_TDM 1
#else
#define HAVE_TDM 0
#endif
#if HAVE_TDM && __has_include(<hip/amd_detail/amd_gfx1250_TDM.h>)
#define TDM_SIX_ARGS 1
#else
#define TDM_SIX_ARGS 0
#endif

// ---------------------------------------------------------------------------
// Farthest point sampling: one block per batch, points cached in LDS,
// per-iteration argmax via LDS tree reduction (first-index tie-break).
// ---------------------------------------------------------------------------
__global__ void fps_kernel(const float* __restrict__ xyz, int n, int npoint,
                           int* __restrict__ outIdx) {
  const int T = 256;
  int b = blockIdx.x;
  int tid = threadIdx.x;
  __shared__ float sx[1024], sy[1024], sz[1024], dist[1024];
  __shared__ float rv[T];
  __shared__ int   ri[T];
  __shared__ int   curFar;

  const float* base = xyz + (size_t)b * n * 3;
  for (int i = tid; i < n; i += T) {
    sx[i] = base[i * 3 + 0];
    sy[i] = base[i * 3 + 1];
    sz[i] = base[i * 3 + 2];
    dist[i] = 1e10f;
  }
  if (tid == 0) curFar = 0;
  __syncthreads();

  for (int it = 0; it < npoint; ++it) {
    int far = curFar;
    if (tid == 0) outIdx[(size_t)b * npoint + it] = far;
    float cx = sx[far], cy = sy[far], cz = sz[far];
    float bestv = -1.0f;
    int   besti = 0;
    for (int i = tid; i < n; i += T) {
      float dx = sx[i] - cx, dy = sy[i] - cy, dz = sz[i] - cz;
      float d = dx * dx + dy * dy + dz * dz;
      float dm = fminf(dist[i], d);
      dist[i] = dm;
      if (dm > bestv) { bestv = dm; besti = i; }
    }
    rv[tid] = bestv; ri[tid] = besti;
    __syncthreads();
    for (int s = T / 2; s > 0; s >>= 1) {
      if (tid < s) {
        if (rv[tid + s] > rv[tid] ||
            (rv[tid + s] == rv[tid] && ri[tid + s] < ri[tid])) {
          rv[tid] = rv[tid + s];
          ri[tid] = ri[tid + s];
        }
      }
      __syncthreads();
    }
    if (tid == 0) curFar = ri[0];
    __syncthreads();
  }
}

__global__ void gather_xyz_kernel(const float* __restrict__ xyz,
                                  const int* __restrict__ idx,
                                  float* __restrict__ out,
                                  int n, int S, int total) {
  int t = blockIdx.x * blockDim.x + threadIdx.x;
  if (t >= total) return;
  int b = t / S;
  int g = idx[t];
  const float* src = xyz + ((size_t)b * n + g) * 3;
  float* dst = out + (size_t)t * 3;
  dst[0] = src[0]; dst[1] = src[1]; dst[2] = src[2];
}

// Ball query: first `ns` ascending indices with d^2 <= r^2; pad with first.
__global__ void query_ball_kernel(const float* __restrict__ xyz,
                                  const float* __restrict__ new_xyz,
                                  int* __restrict__ gidx,
                                  int n, int S, int ns, float r2, int total) {
  int t = blockIdx.x * blockDim.x + threadIdx.x;
  if (t >= total) return;  // total = B*S
  int b = t / S;
  const float* c = new_xyz + (size_t)t * 3;
  float cx = c[0], cy = c[1], cz = c[2];
  const float* base = xyz + (size_t)b * n * 3;
  int* out = gidx + (size_t)t * ns;
  int cnt = 0, first = 0;
  bool found = false;
  for (int i = 0; i < n; ++i) {
    float dx = base[i * 3 + 0] - cx;
    float dy = base[i * 3 + 1] - cy;
    float dz = base[i * 3 + 2] - cz;
    float d2 = dx * dx + dy * dy + dz * dz;
    if (d2 <= r2) {
      if (!found) { found = true; first = i; }
      out[cnt] = i;
      if (++cnt >= ns) break;
    }
  }
  for (int j = cnt; j < ns; ++j) out[j] = first;
}

// Grouped feature matrix (M x Kp) in f16, zero-padded beyond K=3+Cpts.
// gidx==nullptr -> group_all (identity index, no centering).
__global__ void build_feat_kernel(const float* __restrict__ xyz,
                                  const float* __restrict__ new_xyz,
                                  const int* __restrict__ gidx,
                                  const float* __restrict__ points,
                                  f16* __restrict__ out,
                                  int n, int S, int ns, int Cpts, int Kp,
                                  long long total) {
  long long t = (long long)blockIdx.x * blockDim.x + threadIdx.x;
  if (t >= total) return;  // total = M * Kp
  int k = (int)(t % Kp);
  long long m = t / Kp;
  int j = (int)(m % ns);
  long long bs = m / ns;  // b*S + s
  int b = (int)(bs / S);
  int g = gidx ? gidx[bs * ns + j] : j;
  float v = 0.0f;
  if (k < 3) {
    v = xyz[((long long)b * n + g) * 3 + k];
    if (new_xyz) v -= new_xyz[bs * 3 + k];
  } else if (k < 3 + Cpts) {
    v = points[((long long)b * n + g) * Cpts + (k - 3)];
  }
  out[t] = (f16)v;
}

// f32 (rows x K) -> f16 (rows x Kp), zero pad.
__global__ void cvt_pad_kernel(const float* __restrict__ src,
                               f16* __restrict__ dst,
                               int K, int Kp, long long total) {
  long long t = (long long)blockIdx.x * blockDim.x + threadIdx.x;
  if (t >= total) return;  // total = rows * Kp
  int k = (int)(t % Kp);
  long long r = t / Kp;
  dst[t] = (f16)((k < K) ? src[r * K + k] : 0.0f);
}

// ---------------------------------------------------------------------------
// WMMA GEMM: out[M x N] = A[M x Kp] * W[N x Kp]^T, fused bias/BN/ReLU.
// Block = 8 waves = 128(M) x 64(N) macro-tile. Per K-chunk of 32:
//   - W chunk (64 cols x 32 K, 4 KB) staged in LDS, double-buffered: wave 0
//     issues the TDM descriptor for chunk kc+1 while chunk kc is consumed,
//     then waits tensorcnt<=1 (same-wave TDM ops complete in order);
//   - each wave loads one A fragment from global and runs 4 WMMAs against
//     the 4 LDS-resident B fragments (4 accumulators).
// The empty asm with a "memory" clobber after the TDM wait is required: the
// TDM builtin does not reference sW, so without it the compiler sees no
// stores to sW and folds the B-fragment ds_loads to undef.
// M tail (FC head, M=32): rows clamp for loads, stores guarded, no wave
// exits early so per-chunk barriers stay uniform. N tail (fc3, N=40):
// stores guarded per sub-tile.
// ---------------------------------------------------------------------------
__global__ void wmma_gemm_kernel(const f16* __restrict__ A,
                                 const f16* __restrict__ W,
                                 int M, int N, int Kp,
                                 const float* __restrict__ bias,
                                 const float* __restrict__ gamma,
                                 const float* __restrict__ beta,
                                 const float* __restrict__ mean,
                                 const float* __restrict__ var,
                                 int relu,
                                 float* __restrict__ outF,
                                 f16* __restrict__ outH,
                                 int ldOut) {
  __shared__ f16 sW[2][64 * 32];  // sole static LDS alloc (offset 0), 2x4KB

  int tilesN64 = (N + 63) >> 6;
  int tn = blockIdx.x % tilesN64;
  int tm = blockIdx.x / tilesN64;
  int waveId = threadIdx.x >> 5;
  int lane = threadIdx.x & 31;
  int half = lane >> 4;   // 0 or 1
  int l16  = lane & 15;
  int col0 = tn << 6;

  int rowA = tm * 128 + waveId * 16 + l16;           // A row held by this lane
  int rowC = (rowA < M) ? rowA : (M - 1);            // clamp for loads
  const f16* Arow = A + (size_t)rowC * Kp;

  v8f acc[4] = {v8f{}, v8f{}, v8f{}, v8f{}};
  int kChunks = Kp >> 5;

#if HAVE_TDM
  // 2D D#: tile = 64 bytes (8 x 8B units) x 64 rows; row stride = Kp*2 bytes.
  auto tdm_issue = [&](int k0e, unsigned ldsOff) {
    unsigned long long ga = (unsigned long long)(size_t)W +
                            (size_t)col0 * Kp * 2 + (size_t)k0e * 2;
    unsigned tdim0 = (unsigned)(Kp >> 2);            // row length in 8B units
    unsigned nrem  = (unsigned)(N - col0);           // valid rows (OOB -> 0)
    u32x4 g0;
    g0[0] = 1u;                                      // count=1, user descriptor
    g0[1] = ldsOff;                                  // LDS byte offset
    g0[2] = (unsigned)(ga & 0xFFFFFFFFu);
    g0[3] = (unsigned)((ga >> 32) & 0x01FFFFFFu) | (2u << 30);  // type=2
    i32x8 g1;
    g1[0] = (int)(3u << 16);                         // data_size = 8B
    g1[1] = (int)((tdim0 & 0xFFFFu) << 16);          // tensor_dim0 [15:0]
    g1[2] = (int)(((tdim0 >> 16) & 0xFFFFu) | ((nrem & 0xFFFFu) << 16));
    g1[3] = (int)(((nrem >> 16) & 0xFFFFu) | (8u << 16));  // tile_dim0 = 8
    g1[4] = 64;                                      // tile_dim1 = 64 rows
    g1[5] = (int)tdim0;                              // dim0 stride low 32
    g1[6] = 0;
    g1[7] = 0;
    i32x4 z4 = {0, 0, 0, 0};
#if TDM_SIX_ARGS
    i32x8 z8 = {0, 0, 0, 0, 0, 0, 0, 0};
    __builtin_amdgcn_tensor_load_to_lds(g0, g1, z4, z4, z8, 0);
#else
    __builtin_amdgcn_tensor_load_to_lds(g0, g1, z4, z4, 0);
#endif
  };
  if (waveId == 0) tdm_issue(0, 0u);                 // prologue: chunk 0
#endif

  for (int kc = 0; kc < kChunks; ++kc) {
    int cur = kc & 1;
    const f16* sBuf = &sW[cur][0];

#if HAVE_TDM
    if (waveId == 0) {
      if (kc + 1 < kChunks) {
        tdm_issue((kc + 1) << 5, (unsigned)((cur ^ 1) * (64 * 32 * 2)));
        __builtin_amdgcn_s_wait_tensorcnt(1);        // chunk kc landed
      } else {
        __builtin_amdgcn_s_wait_tensorcnt(0);
      }
    }
    // Compile-time visibility barrier: the TDM wrote sW behind the
    // compiler's back; force the B-fragment LDS loads to be emitted.
    asm volatile("" : : "r"(&sW[0][0]) : "memory");
#else
    for (int idx = threadIdx.x; idx < 64 * 32; idx += 256) {
      int c = idx >> 5, kk = idx & 31;
      int gc = col0 + c;
      sW[cur][idx] = (gc < N) ? W[(size_t)gc * Kp + (kc << 5) + kk] : (f16)0.0f;
    }
#endif
    __syncthreads();

    // ---- A fragment (16-bit A 16x32, contiguous k = 16*half + s, same as B) ----
    int k0 = kc << 5;
    const f16* ap = Arow + k0 + half * 16;
    v8h alo = *(const v8h*)(ap);        // K = k0+half*16    .. +7
    v8h ahi = *(const v8h*)(ap + 8);    // K = k0+half*16+8  .. +15
    v16h a;
#pragma unroll
    for (int i = 0; i < 8; ++i) { a[i] = alo[i]; a[i + 8] = ahi[i]; }
    if (kc + 1 < kChunks) __builtin_prefetch(Arow + k0 + 32, 0, 1);

    // ---- 4 back-to-back WMMAs sharing the A fragment ----
#pragma unroll
    for (int s = 0; s < 4; ++s) {
      v16h b = *(const v16h*)(sBuf + (s * 16 + l16) * 32 + half * 16);
      acc[s] = __builtin_amdgcn_wmma_f32_16x16x32_f16(
          false, a, false, b, (short)0, acc[s], false, false);
    }
    __syncthreads();   // all reads of sW[cur] done before it is re-targeted
  }

  // ---- epilogue: bias + BN + ReLU, guarded stores ----
  int rowBase = tm * 128 + waveId * 16 + half * 8;   // VGPR r -> M = rowBase+r
#pragma unroll
  for (int s = 0; s < 4; ++s) {
    int nOut = col0 + s * 16 + l16;
    if (nOut >= N) continue;
    float sc = 1.0f, sh = 0.0f, bi = 0.0f;
    if (gamma) {
      float g = gamma[nOut], bt = beta[nOut];
      float mu = mean[nOut], vv = var[nOut];
      sc = g * rsqrtf(vv + BN_EPS);
      sh = bt - mu * sc;
    }
    if (bias) bi = bias[nOut];
#pragma unroll
    for (int r = 0; r < 8; ++r) {
      int mOut = rowBase + r;
      if (mOut >= M) continue;
      float v = acc[s][r] + bi;
      v = v * sc + sh;
      if (relu) v = fmaxf(v, 0.0f);
      size_t o = (size_t)mOut * ldOut + nOut;
      if (outH) outH[o] = (f16)v; else outF[o] = v;
    }
  }
}

// Max over the nsample axis: in (BS, ns, C) f16 -> out (BS, C) f32.
__global__ void maxpool_kernel(const f16* __restrict__ in,
                               float* __restrict__ out,
                               int ns, int C, long long total) {
  long long t = (long long)blockIdx.x * blockDim.x + threadIdx.x;
  if (t >= total) return;  // total = BS * C
  long long bs = t / C;
  int c = (int)(t % C);
  const f16* p = in + bs * (long long)ns * C;
  float m = -1e30f;
  for (int j = 0; j < ns; ++j) m = fmaxf(m, (float)p[(long long)j * C + c]);
  out[t] = m;
}

__global__ void log_softmax_kernel(const float* __restrict__ logits,
                                   float* __restrict__ out, int rows, int C) {
  int r = blockIdx.x * blockDim.x + threadIdx.x;
  if (r >= rows) return;
  const float* x = logits + (size_t)r * C;
  float mx = -1e30f;
  for (int c = 0; c < C; ++c) mx = fmaxf(mx, x[c]);
  float s = 0.0f;
  for (int c = 0; c < C; ++c) s += __expf(x[c] - mx);
  float lse = mx + __logf(s);
  for (int c = 0; c < C; ++c) out[(size_t)r * C + c] = x[c] - lse;
}

// ---------------------------------------------------------------------------
// Host-side orchestration.
// Input flattening (jax pytree, dict keys sorted):
//  0 xyz, 1 normal, 2..5 bn1{beta,gamma,mean,var}, 6..9 bn2,
//  10 fc1.b, 11 fc1.w, 12 fc2.b, 13 fc2.w, 14 fc3.b, 15 fc3.w,
//  16..30 sa1 {bn.beta,bn.gamma,bn.mean,bn.var,w} x3, 31..45 sa2, 46..60 sa3.
// d_out: [log_softmax (32x40) | l3p (32x1024)] f32.
// ---------------------------------------------------------------------------
extern "C" void kernel_launch(void* const* d_in, const int* in_sizes, int n_in,
                              void* d_out, int out_size, void* d_ws, size_t ws_size,
                              hipStream_t stream) {
  (void)in_sizes; (void)n_in; (void)out_size; (void)ws_size;

  const int B = 32, NPTS = 1024;
  const int S1 = 512, NS1 = 32;
  const int S2 = 128, NS2 = 64;
  const long long M1 = (long long)B * S1 * NS1;   // 524288
  const long long M2 = (long long)B * S2 * NS2;   // 262144
  const long long M3 = (long long)B * 128;        // 4096 (group_all)

  const float* xyz = (const float*)d_in[0];
  float* out = (float*)d_out;
  float* l3p_out = out + 32 * 40;   // second tuple element region

  // --- workspace layout ---
  char* ws = (char*)d_ws;
  size_t off = 0;
  auto alloc = [&](size_t bytes) {
    size_t o = off; off = (off + bytes + 255) & ~(size_t)255; return o;
  };
  int*   fidx1 = (int*)  (ws + alloc((size_t)B * S1 * 4));
  int*   fidx2 = (int*)  (ws + alloc((size_t)B * S2 * 4));
  float* l1x   = (float*)(ws + alloc((size_t)B * S1 * 3 * 4));
  float* l2x   = (float*)(ws + alloc((size_t)B * S2 * 3 * 4));
  int*   gidx1 = (int*)  (ws + alloc((size_t)B * S1 * NS1 * 4));
  int*   gidx2 = (int*)  (ws + alloc((size_t)B * S2 * NS2 * 4));
  float* l1p   = (float*)(ws + alloc((size_t)B * S1 * 128 * 4));
  float* l2p   = (float*)(ws + alloc((size_t)B * S2 * 256 * 4));
  float* logits= (float*)(ws + alloc((size_t)32 * 40 * 4));
  f16*   l3ph  = (f16*)  (ws + alloc((size_t)32 * 1024 * 2));

  // f16 weights, K padded to mult of 32.  {input idx, N, K, Kp}
  struct WSpec { int idx, N, K, Kp; };
  const WSpec wspec[12] = {
    {20,   64,    3,   32}, {25,   64,   64,   64}, {30,  128,   64,   64},
    {35,  128,  131,  160}, {40,  128,  128,  128}, {45,  256,  128,  128},
    {50,  256,  259,  288}, {55,  512,  256,  256}, {60, 1024,  512,  512},
    {11,  512, 1024, 1024}, {13,  256,  512,  512}, {15,   40,  256,  256},
  };
  f16* wbuf[12];
  for (int i = 0; i < 12; ++i)
    wbuf[i] = (f16*)(ws + alloc((size_t)wspec[i].N * wspec[i].Kp * 2));

  // ping-pong f16 activation buffers (worst cases: M2*160 and M1*128 elems)
  f16* bufA = (f16*)(ws + alloc((size_t)(M2 * 160) * 2));
  f16* bufB = (f16*)(ws + alloc((size_t)(M1 * 128) * 2));

  auto grid1d = [](long long total) { return dim3((unsigned)((total + 255) / 256)); };

  // --- convert weights ---
  for (int i = 0; i < 12; ++i) {
    long long tot = (long long)wspec[i].N * wspec[i].Kp;
    cvt_pad_kernel<<<grid1d(tot), 256, 0, stream>>>(
        (const float*)d_in[wspec[i].idx], wbuf[i], wspec[i].K, wspec[i].Kp, tot);
  }

  auto gemm = [&](const f16* A, int wi, long long M, int N, int Kp,
                  const float* bias, int bnBase, int relu,
                  float* oF, f16* oH, int ld) {
    const float *g = nullptr, *bt = nullptr, *mu = nullptr, *vv = nullptr;
    if (bnBase >= 0) {
      bt = (const float*)d_in[bnBase + 0];
      g  = (const float*)d_in[bnBase + 1];
      mu = (const float*)d_in[bnBase + 2];
      vv = (const float*)d_in[bnBase + 3];
    }
    int tilesM128 = (int)((M + 127) >> 7);
    int tilesN64  = (N + 63) >> 6;
    wmma_gemm_kernel<<<dim3(tilesM128 * tilesN64), 256, 0, stream>>>(
        A, wbuf[wi], (int)M, N, Kp, bias, g, bt, mu, vv, relu, oF, oH, ld);
  };

  // ============================ SA1 ============================
  fps_kernel<<<dim3(B), 256, 0, stream>>>(xyz, NPTS, S1, fidx1);
  gather_xyz_kernel<<<grid1d(B * S1), 256, 0, stream>>>(xyz, fidx1, l1x, NPTS, S1, B * S1);
  query_ball_kernel<<<grid1d(B * S1), 256, 0, stream>>>(
      xyz, l1x, gidx1, NPTS, S1, NS1, 0.04f, B * S1);
  build_feat_kernel<<<grid1d(M1 * 32), 256, 0, stream>>>(
      xyz, l1x, gidx1, nullptr, bufA, NPTS, S1, NS1, 0, 32, M1 * 32);
  gemm(bufA, 0, M1,  64, 32, nullptr, 16, 1, nullptr, bufB,  64);
  gemm(bufB, 1, M1,  64, 64, nullptr, 21, 1, nullptr, bufA,  64);
  gemm(bufA, 2, M1, 128, 64, nullptr, 26, 1, nullptr, bufB, 128);
  maxpool_kernel<<<grid1d((long long)B * S1 * 128), 256, 0, stream>>>(
      bufB, l1p, NS1, 128, (long long)B * S1 * 128);

  // ============================ SA2 ============================
  fps_kernel<<<dim3(B), 256, 0, stream>>>(l1x, S1, S2, fidx2);
  gather_xyz_kernel<<<grid1d(B * S2), 256, 0, stream>>>(l1x, fidx2, l2x, S1, S2, B * S2);
  query_ball_kernel<<<grid1d(B * S2), 256, 0, stream>>>(
      l1x, l2x, gidx2, S1, S2, NS2, 0.16f, B * S2);
  build_feat_kernel<<<grid1d(M2 * 160), 256, 0, stream>>>(
      l1x, l2x, gidx2, l1p, bufA, S1, S2, NS2, 128, 160, M2 * 160);
  gemm(bufA, 3, M2, 128, 160, nullptr, 31, 1, nullptr, bufB, 128);
  gemm(bufB, 4, M2, 128, 128, nullptr, 36, 1, nullptr, bufA, 128);
  gemm(bufA, 5, M2, 256, 128, nullptr, 41, 1, nullptr, bufB, 256);
  maxpool_kernel<<<grid1d((long long)B * S2 * 256), 256, 0, stream>>>(
      bufB, l2p, NS2, 256, (long long)B * S2 * 256);

  // ============================ SA3 (group_all) ============================
  build_feat_kernel<<<grid1d(M3 * 288), 256, 0, stream>>>(
      l2x, nullptr, nullptr, l2p, bufA, S2, 1, S2, 256, 288, M3 * 288);
  gemm(bufA, 6, M3,  256, 288, nullptr, 46, 1, nullptr, bufB,  256);
  gemm(bufB, 7, M3,  512, 256, nullptr, 51, 1, nullptr, bufA,  512);
  gemm(bufA, 8, M3, 1024, 512, nullptr, 56, 1, nullptr, bufB, 1024);
  maxpool_kernel<<<grid1d((long long)B * 1024), 256, 0, stream>>>(
      bufB, l3p_out, S2, 1024, (long long)B * 1024);   // l3p straight to d_out

  // ============================ FC head ============================
  cvt_pad_kernel<<<grid1d((long long)32 * 1024), 256, 0, stream>>>(
      l3p_out, l3ph, 1024, 1024, (long long)32 * 1024);
  gemm(l3ph, 9, 32, 512, 1024, (const float*)d_in[10], 2, 1, nullptr, bufA, 512);
  gemm(bufA, 10, 32, 256,  512, (const float*)d_in[12], 6, 1, nullptr, bufB, 256);
  gemm(bufB, 11, 32,  40,  256, (const float*)d_in[14], -1, 0, logits, nullptr, 40);
  log_softmax_kernel<<<dim3(1), 32, 0, stream>>>(logits, out, 32, 40);
}
